// Criterion_64965675319883
// MI455X (gfx1250) — matrix operand order repaired and checked by reading, not compiled, not run on hardware
//
#include <hip/hip_runtime.h>
#include <hip/hip_bf16.h>
#include <math.h>

// ---------------------------------------------------------------------------
// Proxy-anchor rank loss + KD distillation for MI455X (gfx1250, wave32).
// Dominant GEMM (512 x 100000 x 128) runs on v_wmma_f32_16x16x32_bf16 with a
// bf16 hi/lo split (3 WMMAs per K-tile) and a fused streaming logsumexp in
// base-2 domain (branchless, native v_exp_f32).
// Round-2 changes: launch_bounds to kill scratch spills; 4 waves of a WG share
// one B-tile stream (WGP$ dedup -> ~4x less L2 traffic); 2 accumulators.
// ---------------------------------------------------------------------------

typedef __attribute__((ext_vector_type(16))) __bf16 v16bf;
typedef __attribute__((ext_vector_type(8)))  float  v8f;

#define TAUC     4.0f
#define BETAC    0.5f
#define NEPOCHS  150.0f
#define EPSN     1e-12f

#define B_ROWS 512
#define DDIM   128
#define P_ROWS 100000
#define PTILES (P_ROWS/16)   // 6250 column tiles of p
#define BTILES (B_ROWS/16)   // 32 row tiles of b
#define NSPLIT 64            // column splits (grid.y)
#define GWAVES 4             // waves per GEMM workgroup (= row tiles per WG)

#define LOG2E    1.4426950408889634f
#define TWOLOG2E 2.8853900817779268f
#define LN2      0.6931471805599453f
#define MNEG     (-1.0e30f)

struct __align__(4) bf2 { __bf16 a, b; };

// ---------------------------------------------------------------------------
// Normalize rows (x3), emit sum-of-squares, optional f32 copy, and bf16 hi/lo
// in WMMA operand layout. aLayout=1 -> A-matrix 16x32 layout, else B 32x16.
//   A layout: half=kk>>4,     e=kk&15
//   B layout: half=kk>>4,     e=kk&15
// One wave per row; lane holds 4 consecutive dims (two even-aligned pairs,
// each pair shares one VGPR in the operand layout -> 4-byte packed stores).
// ---------------------------------------------------------------------------
__global__ void nrm_kernel(const float* __restrict__ src, int R,
                           __bf16* __restrict__ hi, __bf16* __restrict__ lo,
                           float* __restrict__ sq, float* __restrict__ nrmout,
                           int aLayout)
{
    const int wid  = threadIdx.x >> 5;
    const int lane = threadIdx.x & 31;
    const int r = blockIdx.x * 8 + wid;
    if (r >= R) return;

    const float4 v = *(const float4*)(src + (size_t)r * DDIM + lane * 4);
    float ss = v.x*v.x + v.y*v.y + v.z*v.z + v.w*v.w;
    #pragma unroll
    for (int off = 16; off; off >>= 1) ss += __shfl_xor(ss, off, 32);

    const float scale = 3.0f / fmaxf(sqrtf(ss), EPSN);
    const float4 w = { v.x*scale, v.y*scale, v.z*scale, v.w*scale };
    if (lane == 0) sq[r] = ss * scale * scale;
    if (nrmout) *(float4*)(nrmout + (size_t)r * DDIM + lane * 4) = w;

    const int tileIdx = r >> 4, rin = r & 15;
    const float vals[4] = { w.x, w.y, w.z, w.w };
    #pragma unroll
    for (int p = 0; p < 2; ++p) {
        const int k  = lane * 4 + p * 2;     // even dim index of the pair
        const int kt = k >> 5, kk = k & 31;
        int half, e;
        if (aLayout) { half = kk >> 4;       e = kk & 15; }
        else         { half = kk >> 4;       e = kk & 15; }
        const int ln = 16 * half + rin;
        const size_t idx = (((size_t)(tileIdx * 4 + kt) * 32 + ln) * 16 + e);
        const float x0 = vals[p*2], x1 = vals[p*2 + 1];
        const __bf16 h0 = (__bf16)x0, h1 = (__bf16)x1;
        const __bf16 l0 = (__bf16)(x0 - (float)h0);
        const __bf16 l1 = (__bf16)(x1 - (float)h1);
        ((bf2*)hi)[idx >> 1] = bf2{ h0, h1 };
        ((bf2*)lo)[idx >> 1] = bf2{ l0, l1 };
    }
}

// ---------------------------------------------------------------------------
// Main kernel: b @ p^T via split-bf16 WMMA, fused streaming base-2 logsumexp
// of -d[i,j]*log2(e) over negative proxies (own-class masked via val=-inf).
// Grid (BTILES/GWAVES, NSPLIT); wave w owns row tile blockIdx.x*4+w, and all
// 4 waves walk the SAME column tiles (t = sp, sp+NSPLIT, ...) so their B
// loads coalesce in the WGP cache. No cross-wave combine needed.
// C layout: lane n=lane&15 is column N, VGPR v is row M = v + 8*(lane>>4).
// ---------------------------------------------------------------------------
__global__ __launch_bounds__(GWAVES * 32, 1)
void gemm_lse_kernel(const __bf16* __restrict__ bhi, const __bf16* __restrict__ blo,
                     const __bf16* __restrict__ phi, const __bf16* __restrict__ plo,
                     const float* __restrict__ bb, const float* __restrict__ pp,
                     const int* __restrict__ labels,
                     float* __restrict__ pm, float* __restrict__ ps)
{
    const int w    = threadIdx.x >> 5;
    const int lane = threadIdx.x & 31;
    const int it   = blockIdx.x * GWAVES + w;  // row tile (0..31)
    const int sp   = blockIdx.y;               // column split (0..63)
    const int half = lane >> 4;
    const int n    = lane & 15;

    const v16bf* bhv = (const v16bf*)bhi;
    const v16bf* blv = (const v16bf*)blo;
    v16bf ah[4], al[4];
    #pragma unroll
    for (int kt = 0; kt < 4; ++kt) {
        ah[kt] = bhv[(size_t)(it * 4 + kt) * 32 + lane];
        al[kt] = blv[(size_t)(it * 4 + kt) * 32 + lane];
    }
    int lab[8]; float bb2[8];
    #pragma unroll
    for (int v = 0; v < 8; ++v) {
        const int row = it * 16 + 8 * half + v;
        lab[v] = labels[row];
        bb2[v] = bb[row] * LOG2E;
    }
    float m[8], s[8];
    #pragma unroll
    for (int v = 0; v < 8; ++v) { m[v] = MNEG; s[v] = 0.0f; }

    const v16bf* phv = (const v16bf*)phi + lane;
    const v16bf* plv = (const v16bf*)plo + lane;

    for (int t = sp; t < PTILES; t += NSPLIT) {
        v8f c0 = {}, c1 = {};
        #pragma unroll
        for (int kt = 0; kt < 4; ++kt) {
            const v16bf bh = phv[(size_t)(t * 4 + kt) * 32];
            const v16bf bl = plv[(size_t)(t * 4 + kt) * 32];
            c0 = __builtin_amdgcn_wmma_f32_16x16x32_bf16(false, ah[kt], false, bh, (short)0, c0, false, false);
            c1 = __builtin_amdgcn_wmma_f32_16x16x32_bf16(false, al[kt], false, bh, (short)0, c1, false, false);
            c1 = __builtin_amdgcn_wmma_f32_16x16x32_bf16(false, ah[kt], false, bl, (short)0, c1, false, false);
        }
        const int   j   = t * 16 + n;
        const float pp2 = pp[j] * LOG2E;
        #pragma unroll
        for (int v = 0; v < 8; ++v) {
            const float dot = c0[v] + c1[v];
            float val = fmaf(dot, TWOLOG2E, -(bb2[v] + pp2));  // -d*log2(e)
            val = (j == lab[v]) ? -INFINITY : val;             // mask own class
            const float nm = fmaxf(m[v], val);
            s[v] = s[v] * exp2f(m[v] - nm) + exp2f(val - nm);  // branchless
            m[v] = nm;
        }
    }

    // Butterfly-combine (m,s) across the 16 lanes of each half (rows fixed).
    #pragma unroll
    for (int off = 1; off <= 8; off <<= 1) {
        #pragma unroll
        for (int v = 0; v < 8; ++v) {
            const float m2 = __shfl_xor(m[v], off, 32);
            const float s2 = __shfl_xor(s[v], off, 32);
            const float nm = fmaxf(m[v], m2);
            s[v] = s[v] * exp2f(m[v] - nm) + s2 * exp2f(m2 - nm);
            m[v] = nm;
        }
    }
    if (n == 0) {
        #pragma unroll
        for (int v = 0; v < 8; ++v) {
            const size_t idx = (size_t)(it * 16 + 8 * half + v) * NSPLIT + sp;
            pm[idx] = m[v]; ps[idx] = s[v];
        }
    }
}

// ---------------------------------------------------------------------------
// Exact f32 positive distance per row: d[i, label_i]. One wave per row.
// ---------------------------------------------------------------------------
__global__ void posdist_kernel(const float* __restrict__ bnrm, const float* __restrict__ proxies,
                               const int* __restrict__ labels, const float* __restrict__ bb,
                               const float* __restrict__ pp, float* __restrict__ pos)
{
    const int wid = threadIdx.x >> 5, lane = threadIdx.x & 31;
    const int i = blockIdx.x * 4 + wid;
    if (i >= B_ROWS) return;
    const int labi = labels[i];
    const float4 bv = *(const float4*)(bnrm    + (size_t)i    * DDIM + lane * 4);
    const float4 qv = *(const float4*)(proxies + (size_t)labi * DDIM + lane * 4);
    float dotr = bv.x*qv.x + bv.y*qv.y + bv.z*qv.z + bv.w*qv.w;
    float qq   = qv.x*qv.x + qv.y*qv.y + qv.z*qv.z + qv.w*qv.w;
    #pragma unroll
    for (int off = 16; off; off >>= 1) {
        dotr += __shfl_xor(dotr, off, 32);
        qq   += __shfl_xor(qq,   off, 32);
    }
    if (lane == 0) {
        const float sc = 3.0f / fmaxf(sqrtf(qq), EPSN);
        pos[i] = bb[i] + pp[labi] - 2.0f * sc * dotr;
    }
}

// ---------------------------------------------------------------------------
// KD term: per-row masked log-softmax KL over sim = b@b^T, tsim = t@t^T.
// 134 MFLOP total, arrays L2-resident -> plain VALU. Block per row.
// ---------------------------------------------------------------------------
__global__ void kd_kernel(const float* __restrict__ bnrm, const float* __restrict__ teach,
                          const int* __restrict__ labels, float* __restrict__ kd)
{
    const int i = blockIdx.x;
    const int tid = threadIdx.x;
    __shared__ float bi[DDIM], ti[DDIM];
    __shared__ float red[256];
    if (tid < DDIM) { bi[tid] = bnrm[(size_t)i * DDIM + tid]; ti[tid] = teach[(size_t)i * DDIM + tid]; }
    __syncthreads();
    const int labi = labels[i];

    float sv[2], tv[2];
    #pragma unroll
    for (int h = 0; h < 2; ++h) {
        const int j = tid + h * 256;
        const float4* br = (const float4*)(bnrm  + (size_t)j * DDIM);
        const float4* tr = (const float4*)(teach + (size_t)j * DDIM);
        float sd = 0.0f, td = 0.0f;
        #pragma unroll 4
        for (int k = 0; k < DDIM / 4; ++k) {
            const float4 b4 = br[k], t4 = tr[k];
            const float4 b0 = ((const float4*)bi)[k], t0 = ((const float4*)ti)[k];
            sd += b4.x*b0.x + b4.y*b0.y + b4.z*b0.z + b4.w*b0.w;
            td += t4.x*t0.x + t4.y*t0.y + t4.z*t0.z + t4.w*t0.w;
        }
        const float wgt = (labels[j] == labi ? 1.0f : BETAC) / TAUC;
        sv[h] = sd * wgt; tv[h] = td * wgt;
    }

    auto redmax = [&](float x) -> float {
        red[tid] = x; __syncthreads();
        for (int o = 128; o; o >>= 1) { if (tid < o) red[tid] = fmaxf(red[tid], red[tid + o]); __syncthreads(); }
        const float r = red[0]; __syncthreads(); return r;
    };
    auto redsum = [&](float x) -> float {
        red[tid] = x; __syncthreads();
        for (int o = 128; o; o >>= 1) { if (tid < o) red[tid] += red[tid + o]; __syncthreads(); }
        const float r = red[0]; __syncthreads(); return r;
    };

    const float smax = redmax(fmaxf(sv[0], sv[1]));
    const float tmax = redmax(fmaxf(tv[0], tv[1]));
    const float ssum = redsum(__expf(sv[0] - smax) + __expf(sv[1] - smax));
    const float tsum = redsum(__expf(tv[0] - tmax) + __expf(tv[1] - tmax));
    const float lss = logf(ssum), lts = logf(tsum);
    float klc = 0.0f;
    #pragma unroll
    for (int h = 0; h < 2; ++h) {
        const float logq = tv[h] - tmax - lts;
        const float logp = sv[h] - smax - lss;
        klc += __expf(logq) * (logq - logp);
    }
    const float kls = redsum(klc);
    if (tid == 0) kd[i] = kls;
}

// ---------------------------------------------------------------------------
// Finalize: combine NSPLIT base-2 lse partials per row, means, 3 outputs.
// ---------------------------------------------------------------------------
__global__ void finalize_kernel(const float* __restrict__ pm, const float* __restrict__ ps,
                                const float* __restrict__ pos, const float* __restrict__ kd,
                                const int* __restrict__ epoch, float* __restrict__ out)
{
    const int tid = threadIdx.x;   // one thread per row (512)
    __shared__ float red[B_ROWS];
    float mm = MNEG, ssum = 0.0f;
    for (int k = 0; k < NSPLIT; ++k) {
        const float m2 = pm[(size_t)tid * NSPLIT + k], s2 = ps[(size_t)tid * NSPLIT + k];
        const float nm = fmaxf(mm, m2);
        ssum = ssum * exp2f(mm - nm) + s2 * exp2f(m2 - nm);
        mm = nm;
    }
    const float neg_lse = (mm + log2f(ssum)) * LN2;   // back to natural log
    red[tid] = pos[tid] + neg_lse; __syncthreads();
    for (int o = 256; o; o >>= 1) { if (tid < o) red[tid] += red[tid + o]; __syncthreads(); }
    const float rank = red[0] / (float)B_ROWS; __syncthreads();

    red[tid] = kd[tid]; __syncthreads();
    for (int o = 256; o; o >>= 1) { if (tid < o) red[tid] += red[tid + o]; __syncthreads(); }
    const float kdm = red[0] / (float)B_ROWS;

    if (tid == 0) {
        const float sc = ((float)epoch[0] / NEPOCHS) * 1.0f * TAUC * TAUC;
        out[0] = rank + sc * kdm;
        out[1] = rank;
        out[2] = kdm;
    }
}

// ---------------------------------------------------------------------------
extern "C" void kernel_launch(void* const* d_in, const int* in_sizes, int n_in,
                              void* d_out, int out_size, void* d_ws, size_t ws_size,
                              hipStream_t stream)
{
    (void)in_sizes; (void)n_in; (void)out_size; (void)ws_size;
    const float* batch   = (const float*)d_in[0];
    const float* teacher = (const float*)d_in[1];
    const float* proxies = (const float*)d_in[2];
    const int*   labels  = (const int*)d_in[3];
    const int*   epoch   = (const int*)d_in[4];
    float* out = (float*)d_out;

    unsigned char* base = (unsigned char*)d_ws;
    size_t off = 0;
    auto wsalloc = [&](size_t bytes) -> void* {
        void* p = base + off;
        off += (bytes + 255) & ~(size_t)255;
        return p;
    };
    float*  b_norm = (float*) wsalloc((size_t)B_ROWS * DDIM * 4);
    float*  bb     = (float*) wsalloc((size_t)B_ROWS * 4);
    float*  pp     = (float*) wsalloc((size_t)P_ROWS * 4);
    __bf16* b_hi   = (__bf16*)wsalloc((size_t)BTILES * 4 * 32 * 16 * 2);
    __bf16* b_lo   = (__bf16*)wsalloc((size_t)BTILES * 4 * 32 * 16 * 2);
    __bf16* p_hi   = (__bf16*)wsalloc((size_t)PTILES * 4 * 32 * 16 * 2);
    __bf16* p_lo   = (__bf16*)wsalloc((size_t)PTILES * 4 * 32 * 16 * 2);
    float*  pm     = (float*) wsalloc((size_t)B_ROWS * NSPLIT * 4);
    float*  ps     = (float*) wsalloc((size_t)B_ROWS * NSPLIT * 4);
    float*  pos    = (float*) wsalloc((size_t)B_ROWS * 4);
    float*  kd     = (float*) wsalloc((size_t)B_ROWS * 4);

    nrm_kernel<<<B_ROWS / 8, 256, 0, stream>>>(batch,   B_ROWS, b_hi, b_lo, bb, b_norm, 1);
    nrm_kernel<<<P_ROWS / 8, 256, 0, stream>>>(proxies, P_ROWS, p_hi, p_lo, pp, nullptr, 0);
    gemm_lse_kernel<<<dim3(BTILES / GWAVES, NSPLIT), GWAVES * 32, 0, stream>>>(
        b_hi, b_lo, p_hi, p_lo, bb, pp, labels, pm, ps);
    posdist_kernel<<<B_ROWS / 4, 128, 0, stream>>>(b_norm, proxies, labels, bb, pp, pos);
    kd_kernel<<<B_ROWS, 256, 0, stream>>>(b_norm, teacher, labels, kd);
    finalize_kernel<<<1, B_ROWS, 0, stream>>>(pm, ps, pos, kd, epoch, out);
}
